// ValidationModel_28552942584276
// MI455X (gfx1250) — compile-verified, matrix operand order checked by reading, not run on hardware
//
#include <hip/hip_runtime.h>
#include <hip/hip_bf16.h>
#include <math.h>

typedef _Float16 h16;
typedef __attribute__((ext_vector_type(16))) _Float16 v16h;
typedef __attribute__((ext_vector_type(8)))  _Float16 v8h;
typedef __attribute__((ext_vector_type(8)))  float    v8f;

#define CHUNKSZ   64
#define KEEPN     64
#define NCHUNKS   128
#define WINDOW    50
#define JAMO      64
#define CE_DIM    128
#define JAMO_EMB  126
#define XPAD      640     // CONCAT 638 padded to 640 (multiple of 32)
#define TDIM      512
#define ROWS      4096    // KEEP * CHUNK
#define MEANK     320     // 300 padded to 320
#define CONVK     256     // 2*128
#define CONVM     131072  // ROWS*32 time steps

// ---------------------------------------------------------------------------
// WMMA fragment helpers (16x16x32 f16, wave32 layouts per CDNA5 ISA 7.12.2)
// A: lane holds one row (lane&15); K = kbase + e (e<8) or kbase+16+(e-8),
//    kbase = (lane>>4)*8.  B packed identically (column = lane&15).
// ---------------------------------------------------------------------------
__device__ __forceinline__ v16h frag_from_two(const h16* p0, const h16* p1) {
  v8h lo = *(const v8h*)p0;
  v8h hi = *(const v8h*)p1;
  v16h a;
#pragma unroll
  for (int i = 0; i < 8; ++i) { a[i] = lo[i]; a[i + 8] = hi[i]; }
  return a;
}

// ---------------------------------------------------------------------------
// Prep kernels
// ---------------------------------------------------------------------------
__global__ void k_f32_to_f16(const float* __restrict__ src, h16* __restrict__ dst, int n) {
  int i = blockIdx.x * blockDim.x + threadIdx.x;
  if (i < n) dst[i] = (h16)src[i];
}

__global__ void k_zero_f32(float* __restrict__ p, int n) {
  int i = blockIdx.x * blockDim.x + threadIdx.x;
  if (i < n) p[i] = 0.0f;
}

// Pack B [K,N] row-major f32 into WMMA B-fragment layout:
// dst[((ntile*nsteps + step)*32 + lane)*16 + e]
__global__ void k_pack_b(const float* __restrict__ src, h16* __restrict__ dst,
                         int Kvalid, int Nvalid, int nsteps) {
  int lane = threadIdx.x & 31;
  int e    = threadIdx.x >> 5;           // 0..15
  int ntile = blockIdx.x, step = blockIdx.y;
  int kbase = (lane >> 4) * 8;
  int k = step * 32 + kbase + e + ((e >= 8) ? 8 : 0);
  int n = ntile * 16 + (lane & 15);
  float v = 0.0f;
  if (k < Kvalid && n < Nvalid) v = src[(size_t)k * Nvalid + n];
  dst[(((size_t)ntile * nsteps + step) * 32 + lane) * 16 + e] = (h16)v;
}

// ---------------------------------------------------------------------------
// Stream compaction (faithful int32-wraparound chunk sums + first-64 scan)
// ---------------------------------------------------------------------------
__global__ void k_flags(const unsigned* __restrict__ t0, const unsigned* __restrict__ t1,
                        const unsigned* __restrict__ t2, const unsigned* __restrict__ t3,
                        const unsigned* __restrict__ t4, int* __restrict__ flags) {
  int ty = blockIdx.y;
  const unsigned* base = (ty == 0) ? t0 : (ty == 1) ? t1 : (ty == 2) ? t2 : (ty == 3) ? t3 : t4;
  int s = (ty == 0) ? JAMO : WINDOW;
  int n = CHUNKSZ * s;
  const unsigned* p = base + (size_t)blockIdx.x * n;
  unsigned acc = 0;
  for (int i = threadIdx.x; i < n; i += 256) acc += p[i];
  __shared__ unsigned red[256];
  red[threadIdx.x] = acc;
  __syncthreads();
  for (int st = 128; st > 0; st >>= 1) {
    if (threadIdx.x < st) red[threadIdx.x] += red[threadIdx.x + st];
    __syncthreads();
  }
  if (threadIdx.x == 0) flags[ty * NCHUNKS + blockIdx.x] = (red[0] != 0u) ? 1 : 0;
}

__global__ void k_compact(const int* __restrict__ flags, int* __restrict__ comp) {
  if (threadIdx.x != 0) return;
  int t = blockIdx.x;
  int cnt = 0;
  for (int c = 0; c < NCHUNKS && cnt < KEEPN; ++c)
    if (flags[t * NCHUNKS + c]) comp[t * KEEPN + cnt++] = c;
  while (cnt < KEEPN) comp[t * KEEPN + cnt++] = 0;
}

// ---------------------------------------------------------------------------
// Gather + mean-pool (4 ctx tensors), emit f16 means padded to K=320.
// Memory-bound stage: global_prefetch_b8 four index-rows ahead to hide HBM
// latency behind the running accumulation.
// ---------------------------------------------------------------------------
__global__ void k_gather_mean(const int* __restrict__ i0, const int* __restrict__ i1,
                              const int* __restrict__ i2, const int* __restrict__ i3,
                              const float* __restrict__ we, const float* __restrict__ ee,
                              const int* __restrict__ comp, h16* __restrict__ amean) {
  int t = blockIdx.y;   // 0..3 -> inputs 1..4, comp slots 1..4
  int row = blockIdx.x; // 0..4095
  int chunk = row >> 6;
  int src_chunk = comp[(t + 1) * KEEPN + chunk];
  int srow = src_chunk * CHUNKSZ + (row & 63);
  const int* idxp = ((t == 0) ? i0 : (t == 1) ? i1 : (t == 2) ? i2 : i3) + (size_t)srow * WINDOW;
  __shared__ int sidx[WINDOW];
  if (threadIdx.x < WINDOW) sidx[threadIdx.x] = idxp[threadIdx.x];
  __syncthreads();
  const float* table = (t < 2) ? we : ee;
  int col = threadIdx.x;   // blockDim = 320
  float s = 0.0f;
  if (col < 300) {
    __builtin_prefetch(table + (size_t)sidx[0] * 300 + col, 0, 1);
    __builtin_prefetch(table + (size_t)sidx[1] * 300 + col, 0, 1);
    __builtin_prefetch(table + (size_t)sidx[2] * 300 + col, 0, 1);
    __builtin_prefetch(table + (size_t)sidx[3] * 300 + col, 0, 1);
    for (int i = 0; i < WINDOW; ++i) {
      if (i + 4 < WINDOW)
        __builtin_prefetch(table + (size_t)sidx[i + 4] * 300 + col, 0, 1);
      s += table[(size_t)sidx[i] * 300 + col];
    }
    s *= (1.0f / WINDOW);
  }
  amean[((size_t)t * ROWS + row) * MEANK + col] = (col < 300) ? (h16)s : (h16)0.0f;
}

// ---------------------------------------------------------------------------
// Char-CNN as GEMM: M=131072, K=256, N=128(126 valid).
// One wave computes a 16x64 tile (4 accumulators) so each A fragment feeds
// 4 WMMAs. Fused bias + ReLU + max-over-time via monotone int atomicMax.
// ---------------------------------------------------------------------------
__global__ void __launch_bounds__(32)
k_conv_gemm(const int* __restrict__ ji, const int* __restrict__ comp,
            const h16* __restrict__ ce16, const h16* __restrict__ bp,
            const float* __restrict__ conv_b, float* __restrict__ cemb) {
  int lane = threadIdx.x;
  int l15 = lane & 15, kbase = (lane >> 4) * 8;
  int mrow = blockIdx.x * 16 + l15;          // conv-output row (r*32 + t)
  int r = mrow >> 5, tt = mrow & 31;
  int srow = comp[(r >> 6)] * CHUNKSZ + (r & 63);   // slot 0
  int jv0 = ji[(size_t)srow * JAMO + 2 * tt];
  int jv1 = ji[(size_t)srow * JAMO + 2 * tt + 1];
  v8f acc[4] = {};
#pragma unroll
  for (int s = 0; s < 8; ++s) {
    int jv = (s < 4) ? jv0 : jv1;
    int chan = (s & 3) * 32 + kbase;
    const h16* ap = ce16 + (size_t)jv * CE_DIM + chan;
    v16h a = frag_from_two(ap, ap + 16);
#pragma unroll
    for (int j = 0; j < 4; ++j) {
      int nt = blockIdx.y * 4 + j;
      v16h b = *(const v16h*)(bp + (((size_t)nt * 8 + s) * 32 + lane) * 16);
      acc[j] = __builtin_amdgcn_wmma_f32_16x16x32_f16(false, a, false, b, (short)0, acc[j], false, false);
    }
  }
#pragma unroll
  for (int j = 0; j < 4; ++j) {
    int n = (blockIdx.y * 4 + j) * 16 + l15;
    if (n < JAMO_EMB) {
      float bias = conv_b[n];
      float m = 0.0f;
#pragma unroll
      for (int v = 0; v < 8; ++v) {
        float x = acc[j][v] + bias;
        x = x > 0.0f ? x : 0.0f;
        m = m > x ? m : x;
      }
      atomicMax((int*)&cemb[(size_t)r * CE_DIM + n], __float_as_int(m));
    }
  }
}

// ---------------------------------------------------------------------------
// Four 4096x320x128 projection GEMMs, 16x64 tile per wave (4 accumulators)
// ---------------------------------------------------------------------------
__global__ void __launch_bounds__(32)
k_proj_gemm(const h16* __restrict__ amean, const h16* __restrict__ bpctx,
            float* __restrict__ proj) {
  int lane = threadIdx.x;
  int l15 = lane & 15, kbase = (lane >> 4) * 8;
  int tz = blockIdx.z;
  int mrow = blockIdx.x * 16 + l15;
  const h16* arow = amean + ((size_t)tz * ROWS + mrow) * MEANK;
  const h16* bpt  = bpctx + (size_t)tz * (8 * 10 * 512);
  v8f acc[4] = {};
#pragma unroll
  for (int s = 0; s < 10; ++s) {
    const h16* ap = arow + s * 32 + kbase;
    v16h a = frag_from_two(ap, ap + 16);
#pragma unroll
    for (int j = 0; j < 4; ++j) {
      int nt = blockIdx.y * 4 + j;
      v16h b = *(const v16h*)(bpt + (((size_t)nt * 10 + s) * 32 + lane) * 16);
      acc[j] = __builtin_amdgcn_wmma_f32_16x16x32_f16(false, a, false, b, (short)0, acc[j], false, false);
    }
  }
  int mb = blockIdx.x * 16 + (lane >> 4) * 8;
#pragma unroll
  for (int j = 0; j < 4; ++j) {
    int n = (blockIdx.y * 4 + j) * 16 + l15;
#pragma unroll
    for (int v = 0; v < 8; ++v)
      proj[((size_t)tz * ROWS + mb + v) * CE_DIM + n] = acc[j][v];
  }
}

// ---------------------------------------------------------------------------
// Build x[4096,640] f16: concat(c_emb 126, tanh(w proj+b) 256, tanh(e proj+b) 256, pad 2)
// ---------------------------------------------------------------------------
__global__ void k_pack_x(const float* __restrict__ cemb, const float* __restrict__ proj,
                         const float* __restrict__ cw_b, const float* __restrict__ ce_b,
                         h16* __restrict__ x) {
  int row = blockIdx.x, col = threadIdx.x;   // blockDim = 640
  float v;
  if (col < 126)       v = cemb[(size_t)row * CE_DIM + col];
  else if (col < 254)  { int j = col - 126; v = tanhf(proj[((size_t)0 * ROWS + row) * CE_DIM + j] + cw_b[j]); }
  else if (col < 382)  { int j = col - 254; v = tanhf(proj[((size_t)1 * ROWS + row) * CE_DIM + j] + cw_b[128 + j]); }
  else if (col < 510)  { int j = col - 382; v = tanhf(proj[((size_t)2 * ROWS + row) * CE_DIM + j] + ce_b[j]); }
  else if (col < 638)  { int j = col - 510; v = tanhf(proj[((size_t)3 * ROWS + row) * CE_DIM + j] + ce_b[128 + j]); }
  else                 v = 0.0f;
  x[(size_t)row * XPAD + col] = (h16)v;
}

// ---------------------------------------------------------------------------
// Transform GEMM: 4096x640x512, 16x64 tile per wave, fused +t_b and ReLU
// ---------------------------------------------------------------------------
__global__ void __launch_bounds__(32)
k_t_gemm(const h16* __restrict__ x, const h16* __restrict__ bpt_all,
         const float* __restrict__ t_b, float* __restrict__ y) {
  int lane = threadIdx.x;
  int l15 = lane & 15, kbase = (lane >> 4) * 8;
  int mrow = blockIdx.x * 16 + l15;
  const h16* arow = x + (size_t)mrow * XPAD;
  v8f acc[4] = {};
#pragma unroll
  for (int s = 0; s < 20; ++s) {
    const h16* ap = arow + s * 32 + kbase;
    v16h a = frag_from_two(ap, ap + 16);
#pragma unroll
    for (int j = 0; j < 4; ++j) {
      int nt = blockIdx.y * 4 + j;
      v16h b = *(const v16h*)(bpt_all + (((size_t)nt * 20 + s) * 32 + lane) * 16);
      acc[j] = __builtin_amdgcn_wmma_f32_16x16x32_f16(false, a, false, b, (short)0, acc[j], false, false);
    }
  }
  int mb = blockIdx.x * 16 + (lane >> 4) * 8;
#pragma unroll
  for (int j = 0; j < 4; ++j) {
    int n = (blockIdx.y * 4 + j) * 16 + l15;
    float bias = t_b[n];
#pragma unroll
    for (int v = 0; v < 8; ++v) {
      float r = acc[j][v] + bias;
      y[((size_t)mb + v) * TDIM + n] = r > 0.0f ? r : 0.0f;
    }
  }
}

// Deterministic ordered mean over each 64-row chunk: y[4096,512] -> rep[64,512]
__global__ void k_chunk_mean(const float* __restrict__ y, float* __restrict__ rep) {
  int chunk = blockIdx.x;
  int col = blockIdx.y * 256 + threadIdx.x;   // 0..511
  float s = 0.0f;
  for (int i = 0; i < CHUNKSZ; ++i)
    s += y[((size_t)chunk * CHUNKSZ + i) * TDIM + col];
  rep[(size_t)chunk * TDIM + col] = s * (1.0f / CHUNKSZ);
}

// Head: rep[64,512] @ p1_w[512,100] +b -> ReLU -> @ p2_w[100,1] +b -> sigmoid
__global__ void k_head(const float* __restrict__ rep,
                       const float* __restrict__ p1_w, const float* __restrict__ p1_b,
                       const float* __restrict__ p2_w, const float* __restrict__ p2_b,
                       float* __restrict__ out) {
  int row = blockIdx.x;
  __shared__ float h[100];
  int t = threadIdx.x;   // blockDim = 128
  if (t < 100) {
    float acc = p1_b[t];
    const float* rp = rep + (size_t)row * TDIM;
    for (int k = 0; k < TDIM; ++k) acc += rp[k] * p1_w[(size_t)k * 100 + t];
    h[t] = acc > 0.0f ? acc : 0.0f;
  }
  __syncthreads();
  if (t == 0) {
    float s = p2_b[0];
    for (int j = 0; j < 100; ++j) s += h[j] * p2_w[j];
    out[row] = 1.0f / (1.0f + expf(-s));
  }
}

// ---------------------------------------------------------------------------
extern "C" void kernel_launch(void* const* d_in, const int* in_sizes, int n_in,
                              void* d_out, int out_size, void* d_ws, size_t ws_size,
                              hipStream_t stream) {
  (void)in_sizes; (void)n_in; (void)out_size; (void)ws_size;
  const int*   ji     = (const int*)d_in[0];
  const int*   cwl    = (const int*)d_in[1];
  const int*   cwr    = (const int*)d_in[2];
  const int*   cel    = (const int*)d_in[3];
  const int*   cer    = (const int*)d_in[4];
  const float* we     = (const float*)d_in[6];
  const float* ee     = (const float*)d_in[7];
  const float* cetab  = (const float*)d_in[8];
  const float* conv_w = (const float*)d_in[9];   // [2,128,126] == [256,126] row-major
  const float* conv_b = (const float*)d_in[10];
  const float* wl_w   = (const float*)d_in[11];
  const float* wr_w   = (const float*)d_in[12];
  const float* cw_b   = (const float*)d_in[13];
  const float* el_w   = (const float*)d_in[14];
  const float* er_w   = (const float*)d_in[15];
  const float* ce_b   = (const float*)d_in[16];
  const float* t_w    = (const float*)d_in[17];
  const float* t_b    = (const float*)d_in[18];
  const float* p1_w   = (const float*)d_in[19];
  const float* p1_b   = (const float*)d_in[20];
  const float* p2_w   = (const float*)d_in[21];
  const float* p2_b   = (const float*)d_in[22];

  char*  ws  = (char*)d_ws;
  size_t off = 0;
  auto carve = [&](size_t bytes) -> char* {
    char* p = ws + off;
    off += (bytes + 255) & ~(size_t)255;
    return p;
  };
  int*   comp    = (int*)  carve(5 * KEEPN * 4);
  int*   flags   = (int*)  carve(5 * NCHUNKS * 4);
  h16*   ce16    = (h16*)  carve((size_t)100 * CE_DIM * 2);
  h16*   bp_conv = (h16*)  carve((size_t)8 * 8 * 512 * 2);
  h16*   bp_t    = (h16*)  carve((size_t)32 * 20 * 512 * 2);
  h16*   bp_ctx  = (h16*)  carve((size_t)4 * 8 * 10 * 512 * 2);
  h16*   amean   = (h16*)  carve((size_t)4 * ROWS * MEANK * 2);
  float* cemb    = (float*)carve((size_t)ROWS * CE_DIM * 4);
  float* proj    = (float*)carve((size_t)4 * ROWS * CE_DIM * 4);
  h16*   xbuf    = (h16*)  carve((size_t)ROWS * XPAD * 2);
  float* ybuf    = (float*)carve((size_t)ROWS * TDIM * 4);
  float* rep     = (float*)carve((size_t)KEEPN * TDIM * 4);

  // Prep: f16 char table + packed B fragments
  k_f32_to_f16<<<(100 * CE_DIM + 255) / 256, 256, 0, stream>>>(cetab, ce16, 100 * CE_DIM);
  k_pack_b<<<dim3(8, 8),   512, 0, stream>>>(conv_w, bp_conv, CONVK, JAMO_EMB, 8);
  k_pack_b<<<dim3(32, 20), 512, 0, stream>>>(t_w, bp_t, 638, TDIM, 20);
  k_pack_b<<<dim3(8, 10),  512, 0, stream>>>(wl_w, bp_ctx + (size_t)0 * 8 * 10 * 512, 300, 128, 10);
  k_pack_b<<<dim3(8, 10),  512, 0, stream>>>(wr_w, bp_ctx + (size_t)1 * 8 * 10 * 512, 300, 128, 10);
  k_pack_b<<<dim3(8, 10),  512, 0, stream>>>(el_w, bp_ctx + (size_t)2 * 8 * 10 * 512, 300, 128, 10);
  k_pack_b<<<dim3(8, 10),  512, 0, stream>>>(er_w, bp_ctx + (size_t)3 * 8 * 10 * 512, 300, 128, 10);

  // Stream compaction
  k_flags<<<dim3(NCHUNKS, 5), 256, 0, stream>>>((const unsigned*)ji, (const unsigned*)cwl,
                                                (const unsigned*)cwr, (const unsigned*)cel,
                                                (const unsigned*)cer, flags);
  k_compact<<<5, 32, 0, stream>>>(flags, comp);

  // Gather-mean (memory-bound stage, prefetch-pipelined) + conv GEMM (WMMA)
  k_zero_f32<<<(ROWS * CE_DIM + 255) / 256, 256, 0, stream>>>(cemb, ROWS * CE_DIM);
  k_gather_mean<<<dim3(ROWS, 4), 320, 0, stream>>>(cwl, cwr, cel, cer, we, ee, comp, amean);
  k_conv_gemm<<<dim3(CONVM / 16, CE_DIM / 64), 32, 0, stream>>>(ji, comp, ce16, bp_conv, conv_b, cemb);

  // Context projections (WMMA) + tanh/pack + transform GEMM (WMMA)
  k_proj_gemm<<<dim3(ROWS / 16, CE_DIM / 64, 4), 32, 0, stream>>>(amean, bp_ctx, proj);
  k_pack_x<<<ROWS, XPAD, 0, stream>>>(cemb, proj, cw_b, ce_b, xbuf);
  k_t_gemm<<<dim3(ROWS / 16, TDIM / 64), 32, 0, stream>>>(xbuf, bp_t, t_b, ybuf);

  // Mean pool + head
  k_chunk_mean<<<dim3(KEEPN, 2), 256, 0, stream>>>(ybuf, rep);
  k_head<<<KEEPN, 128, 0, stream>>>(rep, p1_w, p1_b, p2_w, p2_b, (float*)d_out);
}
